// IntegratedAttention_63127429317238
// MI455X (gfx1250) — matrix operand order repaired and checked by reading, not compiled, not run on hardware
//
#include <hip/hip_runtime.h>
#include <hip/hip_bf16.h>

// ---------------------------------------------------------------------------
// IntegratedAttention for MI455X (gfx1250, wave32, WMMA f16 16x16x32)
// ---------------------------------------------------------------------------
// Shapes: B=2, S=1024, D=1024, H=16, hd=64.
// All GEMMs and both attention paths run on v_wmma_f32_16x16x32_f16.
// GEMM tiles are staged with gfx1250 async global->LDS DMA (ASYNCcnt),
// double-buffered so staging of tile i+1 overlaps WMMA on tile i.
// Attention is flash-style (online softmax) so the 128MB score matrices
// never touch HBM; the local path truncates its KV loop at eff.
// ---------------------------------------------------------------------------

#define DIMS 1024
#define SEQ  1024
#define NB   2
#define HEADS 16
#define HD   64

typedef __attribute__((ext_vector_type(16))) _Float16 v16h;
typedef __attribute__((ext_vector_type(8)))  float    v8f;

union Frag16 {          // one WMMA 16-bit operand: 16 halfs per lane = 2x uint4
    v16h v;
    uint4 q[2];
};

__device__ __forceinline__ v8f wmma_f16(const v16h& a, const v16h& b, const v8f& c) {
    return __builtin_amdgcn_wmma_f32_16x16x32_f16(false, a, false, b, (short)0, c,
                                                  false, false);
}

// gfx1250 async global->LDS copy, 16B per lane, tracked by ASYNCcnt.
// VDST operand is the per-lane LDS byte address (addrspace(3) offset).
__device__ __forceinline__ void async_copy_b128(const _Float16* lds_dst,
                                                const _Float16* gsrc) {
    const unsigned loff =
        (unsigned)(size_t)(__attribute__((address_space(3))) const _Float16*)lds_dst;
    asm volatile("global_load_async_to_lds_b128 %0, %1, off"
                 :: "v"(loff), "v"(gsrc) : "memory");
}

__device__ __forceinline__ void wait_async0() {
    asm volatile("s_wait_asynccnt 0" ::: "memory");
}

// ---------------------------------------------------------------------------
// Fused dual LayerNorm: x -> f16 local (ln_a) and f16 globe (ln_b)
// one block per row (2048 rows), 256 threads
// ---------------------------------------------------------------------------
__global__ __launch_bounds__(256) void ln_dual_kernel(
    const float* __restrict__ x,
    const float* __restrict__ ga, const float* __restrict__ ba,
    const float* __restrict__ gb, const float* __restrict__ bb,
    _Float16* __restrict__ outA, _Float16* __restrict__ outB)
{
    __shared__ float red[256];
    const int row = blockIdx.x;
    const int tid = threadIdx.x;
    const float* xr = x + (size_t)row * DIMS;

    float v[4];
    float s = 0.0f;
#pragma unroll
    for (int i = 0; i < 4; i++) { v[i] = xr[tid + i * 256]; s += v[i]; }

    red[tid] = s; __syncthreads();
    for (int o = 128; o > 0; o >>= 1) {
        if (tid < o) red[tid] += red[tid + o];
        __syncthreads();
    }
    const float mean = red[0] * (1.0f / DIMS);
    __syncthreads();

    float s2 = 0.0f;
#pragma unroll
    for (int i = 0; i < 4; i++) { float d = v[i] - mean; s2 += d * d; }
    red[tid] = s2; __syncthreads();
    for (int o = 128; o > 0; o >>= 1) {
        if (tid < o) red[tid] += red[tid + o];
        __syncthreads();
    }
    const float rstd = rsqrtf(red[0] * (1.0f / DIMS) + 1e-5f);

#pragma unroll
    for (int i = 0; i < 4; i++) {
        const int c = tid + i * 256;
        const float n = (v[i] - mean) * rstd;
        outA[(size_t)row * DIMS + c] = (_Float16)(n * ga[c] + ba[c]);
        outB[(size_t)row * DIMS + c] = (_Float16)(n * gb[c] + bb[c]);
    }
}

// ---------------------------------------------------------------------------
// f32 -> f16 converter (weights)
// ---------------------------------------------------------------------------
__global__ __launch_bounds__(256) void cvt_f16_kernel(const float* __restrict__ in,
                                                      _Float16* __restrict__ out, int n)
{
    const int i = blockIdx.x * 256 + threadIdx.x;
    if (i < n) out[i] = (_Float16)in[i];
}

// ---------------------------------------------------------------------------
// Tiled WMMA GEMM: C[M,N] = A[M,K] @ W[N,K]^T + bias
//   block = 256 threads (8 waves); tile 128(M) x 128(N), K-step 32
//   double-buffered LDS staged by async global->LDS DMA (no VGPR round-trip);
//   one barrier per K-step; each wave computes a 16x128 strip with all 8
//   B-fragments preloaded so the 8 WMMAs issue back-to-back.
//   LDS tiles padded to 40 halfs/row (80B) => 16B-aligned fragment reads.
// Optional f32 output (ld = N) and/or f16 output at (ldH, colOffH).
// ---------------------------------------------------------------------------
__global__ __launch_bounds__(256) void gemm_wmma_kernel(
    const _Float16* __restrict__ A, const _Float16* __restrict__ W,
    const float* __restrict__ bias,
    float* __restrict__ outF, _Float16* __restrict__ outH,
    int M, int N, int K, int ldH, int colOffH)
{
    __shared__ _Float16 ldsA[2][128 * 40];
    __shared__ _Float16 ldsW[2][128 * 40];

    const int tid  = threadIdx.x;
    const int wave = tid >> 5;
    const int lane = tid & 31;
    const int g    = lane >> 4;   // lane half (K split)
    const int ln   = lane & 15;   // row (A) / col (B,C)

    const int rowBase = blockIdx.y * 128;
    const int colBase = blockIdx.x * 128;

    v8f acc[8];
#pragma unroll
    for (int t = 0; t < 8; t++)
#pragma unroll
        for (int i = 0; i < 8; i++) acc[t][i] = 0.0f;

    const int lr = tid >> 1;          // 0..127: tile row staged by this thread
    const int lc = (tid & 1) * 16;    // 0 or 16: K offset staged

    const _Float16* gA = A + (size_t)(rowBase + lr) * K + lc;
    const _Float16* gW = W + (size_t)(colBase + lr) * K + lc;
    const int ldsIdx = lr * 40 + lc;

    // prologue: stage K-step 0 into buffer 0
    async_copy_b128(&ldsA[0][ldsIdx],     gA);
    async_copy_b128(&ldsA[0][ldsIdx + 8], gA + 8);
    async_copy_b128(&ldsW[0][ldsIdx],     gW);
    async_copy_b128(&ldsW[0][ldsIdx + 8], gW + 8);
    wait_async0();
    __syncthreads();

    const int nSteps = K >> 5;
    for (int s = 0; s < nSteps; s++) {
        const int buf = s & 1;
        if (s + 1 < nSteps) {   // stage next K-step into the other buffer
            const int kb = (s + 1) << 5;
            async_copy_b128(&ldsA[buf ^ 1][ldsIdx],     gA + kb);
            async_copy_b128(&ldsA[buf ^ 1][ldsIdx + 8], gA + kb + 8);
            async_copy_b128(&ldsW[buf ^ 1][ldsIdx],     gW + kb);
            async_copy_b128(&ldsW[buf ^ 1][ldsIdx + 8], gW + kb + 8);
        }

        Frag16 af;  // A: lane ln = row; j<8 -> k=g*8+j ; j>=8 -> k=16+g*8+(j-8)
        af.q[0] = *(const uint4*)&ldsA[buf][(wave * 16 + ln) * 40 + g * 8];
        af.q[1] = *(const uint4*)&ldsA[buf][(wave * 16 + ln) * 40 + 16 + g * 8];
        Frag16 bf[8];
#pragma unroll
        for (int nt = 0; nt < 8; nt++) {   // B: lane ln = col n; j<8 -> k=g*8+j ; j>=8 -> k=16+g*8+(j-8)
            bf[nt].q[0] = *(const uint4*)&ldsW[buf][(nt * 16 + ln) * 40 + g * 8];
            bf[nt].q[1] = *(const uint4*)&ldsW[buf][(nt * 16 + ln) * 40 + 16 + g * 8];
        }
#pragma unroll
        for (int nt = 0; nt < 8; nt++)
            acc[nt] = wmma_f16(af.v, bf[nt].v, acc[nt]);

        wait_async0();      // next buffer fully staged (all waves, after barrier)
        __syncthreads();
    }

#pragma unroll
    for (int nt = 0; nt < 8; nt++) {
        const int col = colBase + nt * 16 + ln;
        const float bv = bias ? bias[col] : 0.0f;
#pragma unroll
        for (int r = 0; r < 8; r++) {
            const int row = rowBase + wave * 16 + r + 8 * g;  // C: M = r + 8g
            const float v = acc[nt][r] + bv;
            if (outF) outF[(size_t)row * N + col] = v;
            if (outH) outH[(size_t)row * ldH + colOffH + col] = (_Float16)v;
        }
    }
}

// ---------------------------------------------------------------------------
// Flash attention (global & local paths).
//   one wave per 16-row Q tile of one (b,h); 4 waves/block; grid = 512
//   KV streamed in 32-col chunks; online softmax in registers
//   P staged f16 in wave-private LDS; V transposed in wave-private LDS
//   span==nullptr -> global (eff=S, scale=1/64); else local:
//     eff = span[0], scale = (hd^-0.25)^2 / temp = 0.125/span[1], rows>=eff zeroed
// Q/K/V all addressed as [B,S,D] with head offset h*64 (so the local path can
// feed the raw layernorm tensor directly).
// ---------------------------------------------------------------------------
__global__ __launch_bounds__(128) void flash_attn_kernel(
    const _Float16* __restrict__ Q, const _Float16* __restrict__ Kt,
    const _Float16* __restrict__ V,
    _Float16* __restrict__ outH, int ldH, int colOff,
    const float* __restrict__ span, int isLocal)
{
    __shared__ _Float16 lds[4 * 2560];   // per wave: Vt[64][32] + P[16][32]

    const int tid  = threadIdx.x;
    const int wave = tid >> 5;
    const int lane = tid & 31;
    const int g    = lane >> 4;
    const int ln   = lane & 15;

    _Float16* Vt = &lds[wave * 2560];          // [d][kv] transposed V chunk
    _Float16* Pb = &lds[wave * 2560 + 2048];   // [m][kv] probabilities

    const int wid = blockIdx.x * 4 + wave;     // 0..2047
    const int b   = wid >> 10;
    const int h   = (wid >> 6) & 15;
    const int qt  = wid & 63;

    int eff = SEQ;
    float scale = 1.0f / 64.0f;                // (hd^-.25)^2 / sqrt(hd)
    if (span) { eff = (int)span[0]; scale = 0.125f / span[1]; }

    Frag16 aq[2];                              // Q tile, hd=64 -> 2 K-steps
    {
        const _Float16* qrow = Q + ((size_t)(b * SEQ + qt * 16 + ln)) * DIMS + h * HD;
#pragma unroll
        for (int ks = 0; ks < 2; ks++) {
            aq[ks].q[0] = *(const uint4*)(qrow + ks * 32 + g * 8);
            aq[ks].q[1] = *(const uint4*)(qrow + ks * 32 + 16 + g * 8);
        }
    }

    v8f acc[4];
#pragma unroll
    for (int nt = 0; nt < 4; nt++)
#pragma unroll
        for (int i = 0; i < 8; i++) acc[nt][i] = 0.0f;
    float m[8], l[8];
#pragma unroll
    for (int r = 0; r < 8; r++) { m[r] = -3.0e38f; l[r] = 0.0f; }

    const int nChunks = (eff + 31) >> 5;
    for (int c = 0; c < nChunks; c++) {
        const int kv0 = c * 32;

        // stage V chunk transposed: lane loads one 64-half V row, scatters cols
        {
            const _Float16* vrow =
                V + ((size_t)(b * SEQ + kv0 + lane)) * DIMS + h * HD;
            uint4 buf[8];
#pragma unroll
            for (int i = 0; i < 8; i++) buf[i] = ((const uint4*)vrow)[i];
            const _Float16* bh = (const _Float16*)buf;
#pragma unroll
            for (int d = 0; d < 64; d++) Vt[d * 32 + lane] = bh[d];
        }

        // ---- scores: two 16x16 tiles over this 32-col chunk ----
        float p[2][8];
        float mt[8];
#pragma unroll
        for (int r = 0; r < 8; r++) mt[r] = -3.0e38f;

#pragma unroll
        for (int t = 0; t < 2; t++) {
            v8f cs;
#pragma unroll
            for (int i = 0; i < 8; i++) cs[i] = 0.0f;
            const _Float16* krow =
                Kt + ((size_t)(b * SEQ + kv0 + t * 16 + ln)) * DIMS + h * HD;
#pragma unroll
            for (int ks = 0; ks < 2; ks++) {
                Frag16 bk;  // B = K^T: lane = kv col, k-dim = head dim
                bk.q[0] = *(const uint4*)(krow + ks * 32 + g * 8);
                bk.q[1] = *(const uint4*)(krow + ks * 32 + 16 + g * 8);
                cs = wmma_f16(aq[ks].v, bk.v, cs);
            }
            const int col = kv0 + t * 16 + ln;
#pragma unroll
            for (int r = 0; r < 8; r++) {
                const float s = (col < eff) ? cs[r] * scale : -3.0e38f;
                p[t][r] = s;
                mt[r] = fmaxf(mt[r], s);
            }
        }

        // row-max across the 16 lanes of this lane-half
#pragma unroll
        for (int off = 1; off < 16; off <<= 1)
#pragma unroll
            for (int r = 0; r < 8; r++)
                mt[r] = fmaxf(mt[r], __shfl_xor(mt[r], off, 32));

        // online softmax update
#pragma unroll
        for (int r = 0; r < 8; r++) {
            const float mn = fmaxf(m[r], mt[r]);
            const float sc = __expf(m[r] - mn);
            m[r] = mn;
            l[r] *= sc;
#pragma unroll
            for (int nt = 0; nt < 4; nt++) acc[nt][r] *= sc;
            const float p0 = __expf(p[0][r] - mn);
            const float p1 = __expf(p[1][r] - mn);
            p[0][r] = p0; p[1][r] = p1;
            float ps = p0 + p1;
#pragma unroll
            for (int off = 1; off < 16; off <<= 1) ps += __shfl_xor(ps, off, 32);
            l[r] += ps;
        }

        // stage P (C-layout -> A-layout via wave-private LDS)
#pragma unroll
        for (int t = 0; t < 2; t++)
#pragma unroll
            for (int r = 0; r < 8; r++)
                Pb[(r + 8 * g) * 32 + t * 16 + ln] = (_Float16)p[t][r];
        asm volatile("s_wait_dscnt 0" ::: "memory");  // wave-private LDS ordering

        // ---- PV: acc += P(16x32) x V(32x64), fragments batched ----
        Frag16 ap;
        ap.q[0] = *(const uint4*)&Pb[ln * 32 + g * 8];
        ap.q[1] = *(const uint4*)&Pb[ln * 32 + 16 + g * 8];
        Frag16 bvv[4];
#pragma unroll
        for (int nt = 0; nt < 4; nt++) {
            bvv[nt].q[0] = *(const uint4*)&Vt[(nt * 16 + ln) * 32 + g * 8];
            bvv[nt].q[1] = *(const uint4*)&Vt[(nt * 16 + ln) * 32 + 16 + g * 8];
        }
#pragma unroll
        for (int nt = 0; nt < 4; nt++)
            acc[nt] = wmma_f16(ap.v, bvv[nt].v, acc[nt]);
    }

    // epilogue: normalize, zero span-padded rows (local), write f16
#pragma unroll
    for (int nt = 0; nt < 4; nt++) {
#pragma unroll
        for (int r = 0; r < 8; r++) {
            const int srow = qt * 16 + r + 8 * g;
            float v = acc[nt][r] / l[r];
            if (isLocal && srow >= eff) v = 0.0f;
            const size_t row = (size_t)b * SEQ + srow;
            outH[row * ldH + colOff + h * HD + nt * 16 + ln] = (_Float16)v;
        }
    }
}

// ---------------------------------------------------------------------------
// span predictor: mean over S of g_out (f32) -> [B,D]
// ---------------------------------------------------------------------------
__global__ __launch_bounds__(256) void colmean_kernel(const float* __restrict__ g,
                                                      float* __restrict__ out)
{
    const int idx = blockIdx.x * 256 + threadIdx.x;   // B*D = 2048
    const int b = idx >> 10;
    const int d = idx & 1023;
    float s = 0.0f;
    for (int t = 0; t < SEQ; t++) s += g[((size_t)b * SEQ + t) * DIMS + d];
    out[idx] = s * (1.0f / SEQ);
}

// sigmoid(gmean @ Wp^T + bp) -> span_mean -> {eff, temp}
__global__ __launch_bounds__(256) void span_fin_kernel(
    const float* __restrict__ gmean, const float* __restrict__ Wp,
    const float* __restrict__ bp, float* __restrict__ params)
{
    __shared__ float red[256];
    __shared__ float sig[NB];
    const int tid = threadIdx.x;
    for (int b = 0; b < NB; b++) {
        float s = 0.0f;
        for (int d = tid; d < DIMS; d += 256) s += gmean[b * DIMS + d] * Wp[d];
        red[tid] = s; __syncthreads();
        for (int o = 128; o > 0; o >>= 1) {
            if (tid < o) red[tid] += red[tid + o];
            __syncthreads();
        }
        if (tid == 0) sig[b] = 1.0f / (1.0f + __expf(-(red[0] + bp[0])));
        __syncthreads();
    }
    if (tid == 0) {
        const float sm = 0.5f * (sig[0] + sig[1]);
        int span_len = (int)floorf(1024.0f * sm);     // MAX_SPAN * mean
        if (span_len > SEQ)  span_len = SEQ;          // min(., S)
        int eff = span_len;
        if (eff > 1024) eff = 1024;                   // min(., MAX_DIST)
        if (eff < 1)    eff = 1;                      // max(., 1)
        params[0] = (float)eff;
        params[1] = 1.0f + 0.01f * (1.0f - sm);       // temperature
    }
}

// ---------------------------------------------------------------------------
extern "C" void kernel_launch(void* const* d_in, const int* in_sizes, int n_in,
                              void* d_out, int out_size, void* d_ws, size_t ws_size,
                              hipStream_t stream)
{
    (void)in_sizes; (void)n_in; (void)out_size; (void)ws_size;

    const float* x     = (const float*)d_in[0];
    const float* ln_ag = (const float*)d_in[1];
    const float* ln_ab = (const float*)d_in[2];
    const float* ln_bg = (const float*)d_in[3];
    const float* ln_bb = (const float*)d_in[4];
    const float* Wq = (const float*)d_in[5];
    const float* bq = (const float*)d_in[6];
    const float* Wk = (const float*)d_in[7];
    const float* Wv = (const float*)d_in[8];
    const float* bv = (const float*)d_in[9];
    const float* Wc = (const float*)d_in[10];
    const float* bc = (const float*)d_in[11];
    const float* Wp = (const float*)d_in[12];
    const float* bp = (const float*)d_in[13];
    const float* Wo = (const float*)d_in[14];
    const float* bo = (const float*)d_in[15];
    float* out = (float*)d_out;

    // workspace carve (256B aligned)
    char* p = (char*)d_ws;
    auto carve = [&](size_t bytes) -> void* {
        void* r = (void*)p;
        p += (bytes + 255) & ~(size_t)255;
        return r;
    };
    const size_t MS = (size_t)NB * SEQ;              // 2048 rows
    _Float16* local_h = (_Float16*)carve(MS * DIMS * 2);
    _Float16* globe_h = (_Float16*)carve(MS * DIMS * 2);
    _Float16* Wq_h = (_Float16*)carve((size_t)DIMS * DIMS * 2);
    _Float16* Wk_h = (_Float16*)carve((size_t)DIMS * DIMS * 2);
    _Float16* Wv_h = (_Float16*)carve((size_t)DIMS * DIMS * 2);
    _Float16* Wc_h = (_Float16*)carve((size_t)DIMS * DIMS * 2);
    _Float16* Wo_h = (_Float16*)carve((size_t)DIMS * 2 * DIMS * 2);
    _Float16* q_h  = (_Float16*)carve(MS * DIMS * 2);
    _Float16* k_h  = (_Float16*)carve(MS * DIMS * 2);
    _Float16* v_h  = (_Float16*)carve(MS * DIMS * 2);
    _Float16* gatt_h   = (_Float16*)carve(MS * DIMS * 2);
    _Float16* concat_h = (_Float16*)carve(MS * 2 * DIMS * 2);
    float* gout_f = (float*)carve(MS * DIMS * 4);
    float* gmean  = (float*)carve((size_t)NB * DIMS * 4);
    float* params = (float*)carve(256);

    // 1) fused dual LayerNorm -> f16
    ln_dual_kernel<<<MS, 256, 0, stream>>>(x, ln_ag, ln_ab, ln_bg, ln_bb,
                                           local_h, globe_h);

    // 2) weights -> f16
    const int wn = DIMS * DIMS;
    cvt_f16_kernel<<<(wn + 255) / 256, 256, 0, stream>>>(Wq, Wq_h, wn);
    cvt_f16_kernel<<<(wn + 255) / 256, 256, 0, stream>>>(Wk, Wk_h, wn);
    cvt_f16_kernel<<<(wn + 255) / 256, 256, 0, stream>>>(Wv, Wv_h, wn);
    cvt_f16_kernel<<<(wn + 255) / 256, 256, 0, stream>>>(Wc, Wc_h, wn);
    cvt_f16_kernel<<<(2 * wn + 255) / 256, 256, 0, stream>>>(Wo, Wo_h, 2 * wn);

    // 3) QKV projections (M=2048, N=1024, K=1024)
    const dim3 ggrid(DIMS / 128, MS / 128);
    gemm_wmma_kernel<<<ggrid, 256, 0, stream>>>(globe_h, Wq_h, bq, nullptr, q_h,
                                                (int)MS, DIMS, DIMS, DIMS, 0);
    gemm_wmma_kernel<<<ggrid, 256, 0, stream>>>(globe_h, Wk_h, nullptr, nullptr, k_h,
                                                (int)MS, DIMS, DIMS, DIMS, 0);
    gemm_wmma_kernel<<<ggrid, 256, 0, stream>>>(globe_h, Wv_h, bv, nullptr, v_h,
                                                (int)MS, DIMS, DIMS, DIMS, 0);

    // 4) global flash attention (scale = 1/64, full span)
    flash_attn_kernel<<<512, 128, 0, stream>>>(q_h, k_h, v_h, gatt_h,
                                               DIMS, 0, nullptr, 0);

    // 5) Wc projection -> g_out f32 (for span predictor) + f16 into concat[:,D:]
    gemm_wmma_kernel<<<ggrid, 256, 0, stream>>>(gatt_h, Wc_h, bc, gout_f, concat_h,
                                                (int)MS, DIMS, DIMS, 2 * DIMS, DIMS);

    // 6) span predictor -> params {eff, temp}
    colmean_kernel<<<(NB * DIMS) / 256, 256, 0, stream>>>(gout_f, gmean);
    span_fin_kernel<<<1, 256, 0, stream>>>(gmean, Wp, bp, params);

    // 7) local flash attention (Q=K=V=local, span-masked) -> concat[:,0:D]
    flash_attn_kernel<<<512, 128, 0, stream>>>(local_h, local_h, local_h, concat_h,
                                               2 * DIMS, 0, params, 1);

    // 8) output projection (M=2048, N=1024, K=2048) -> f32 d_out
    gemm_wmma_kernel<<<ggrid, 256, 0, stream>>>(concat_h, Wo_h, bo, out, nullptr,
                                                (int)MS, DIMS, 2 * DIMS, 0, 0);
}
